// ReLATransformer_2379411882137
// MI455X (gfx1250) — compile-verified, matrix operand order checked by reading, not run on hardware
//
#include <hip/hip_runtime.h>
#include <hip/hip_bf16.h>

// ---------------- model constants ----------------
#define LSEQ   2048
#define DMODEL 1024
#define NHEAD  16
#define DHEAD  64
#define FFDIM  4096
#define NVOCAB 32000
#define NDEPTH 4
#define INNERD (NHEAD * DHEAD)   // 1024

// ---------------- WMMA types ----------------
typedef __bf16 v16bf __attribute__((ext_vector_type(16)));
typedef float  v8f   __attribute__((ext_vector_type(8)));

// ---------------- GEMM tiling ----------------
#define BK  32
#define GT  256            // 8 waves (wave32)
#define LDR (BK + 8)       // padded LDS row stride (halves) for GEMM tiles

// ---------------- gfx1250 async global->LDS staging ----
#if defined(__has_builtin)
#if __has_builtin(__builtin_amdgcn_global_load_async_to_lds_b128) && \
    __has_builtin(__builtin_amdgcn_s_wait_asynccnt)
#define HAVE_ASYNC_LDS 1
#endif
#endif
#ifndef HAVE_ASYNC_LDS
#define HAVE_ASYNC_LDS 0
#endif

typedef int i32x4 __attribute__((vector_size(4 * sizeof(int))));
#define AS_GLOBAL __attribute__((address_space(1)))
#define AS_LDS    __attribute__((address_space(3)))

__device__ __forceinline__ void stage16(const unsigned short* g, unsigned short* l) {
#if HAVE_ASYNC_LDS
  __builtin_amdgcn_global_load_async_to_lds_b128(
      (AS_GLOBAL i32x4*)g, (AS_LDS i32x4*)l, 0, 0);
#else
  *(uint4*)l = *(const uint4*)g;
#endif
}
__device__ __forceinline__ void stage_wait_all() {
#if HAVE_ASYNC_LDS
  __builtin_amdgcn_s_wait_asynccnt(0);
#endif
}

// ---------------- helpers ----------------
__device__ __forceinline__ unsigned short f32_to_bf16(float f) {
  union { float f; unsigned int u; } x; x.f = f;
  unsigned int r = x.u + 0x7FFFu + ((x.u >> 16) & 1u);  // round-to-nearest-even
  return (unsigned short)(r >> 16);
}

__device__ __forceinline__ float gelu_exact(float x) {
  return 0.5f * x * (1.0f + erff(x * 0.70710678118654752f));
}

// 16x32 bf16 fragment from LDS (row-major per lane). CDNA5 16-bit A layout:
// lane L holds row m=L%16; VGPR i holds K pair at
// k = (i>=4?16:0) + (L>=16?8:0) + 2*(i%4)  -> contiguous runs merge to ds_load_b128
__device__ __forceinline__ v16bf load_frag(const unsigned short* p, int ld, int lane) {
  union { v16bf v; unsigned int u[8]; } r;
  const int m  = lane & 15;
  const int kh = (lane & 16) ? 8 : 0;
  const unsigned short* row = p + m * ld;
#pragma unroll
  for (int i = 0; i < 8; ++i) {
    int k = ((i & 4) << 2) + kh + 2 * (i & 3);
    r.u[i] = *(const unsigned int*)(row + k);
  }
  return r.v;
}

// ---------------- WMMA GEMM (full tiles, B is [N,K] row-major, double-buffered) ----
// C = act(A_bf16[M,K](lda) @ B_bf16[N,K](ldb)^T + bias) + resid
//   CbfT : optional transposed bf16 copy of columns >= tc0 (for V^T extraction)
__global__ void __launch_bounds__(GT)
gemm_wmma_kernel(const unsigned short* __restrict__ A, int lda,
                 const unsigned short* __restrict__ B, int ldb,
                 float* __restrict__ C32, unsigned short* __restrict__ Cbf, int ldc,
                 unsigned short* __restrict__ CbfT, int ldcT, int tc0,
                 const float* __restrict__ bias, const float* __restrict__ resid,
                 int M, int N, int K, int do_gelu) {
  constexpr int BM  = 64;
  constexpr int BN  = 128;
  constexpr int ACH = (BM * BK) / (GT * 8);   // 1 async b128 per thread (A)
  constexpr int BCH = (BN * BK) / (GT * 8);   // 2 async b128 per thread (B)
  constexpr int CH  = ACH + BCH;              // per-wave async ops per stage

  __shared__ __align__(16) unsigned short lA[2][BM * LDR];
  __shared__ __align__(16) unsigned short lB[2][BN * LDR];

  const int tid  = threadIdx.x;
  const int lane = tid & 31;
  const int wave = tid >> 5;
  const int wm   = wave >> 2;          // 0..1
  const int wn   = wave & 3;           // 0..3
  const int rowbase = blockIdx.y * BM;
  const int colbase = blockIdx.x * BN;

  auto stage_tiles = [&](int buf, int kb) {
#pragma unroll
    for (int c = 0; c < ACH; ++c) {
      int f8 = tid + c * GT;
      int m = f8 >> 2, ko = (f8 & 3) * 8;
      stage16(A + (size_t)(rowbase + m) * lda + kb + ko, &lA[buf][m * LDR + ko]);
    }
#pragma unroll
    for (int c = 0; c < BCH; ++c) {
      int f8 = tid + c * GT;
      int n = f8 >> 2, ko = (f8 & 3) * 8;
      stage16(B + (size_t)(colbase + n) * ldb + kb + ko, &lB[buf][n * LDR + ko]);
    }
  };

  v8f acc[2][2] = {};

  stage_tiles(0, 0);                    // prologue: fill buffer 0
  int p = 0;
  for (int kb = 0; kb < K; kb += BK, p ^= 1) {
    const bool more = (kb + BK < K);
    if (more) stage_tiles(p ^ 1, kb + BK);   // next slab in flight
#if HAVE_ASYNC_LDS
    if (more) __builtin_amdgcn_s_wait_asynccnt(CH);  // retire current slab only
    else      __builtin_amdgcn_s_wait_asynccnt(0);
#endif
    __syncthreads();

    if (more)  // speculative prefetch of slab after next
      __builtin_prefetch(A + (size_t)(rowbase + (tid & (BM - 1))) * lda + kb + 2 * BK, 0, 3);

    v16bf afr[2], bfr[2];
    afr[0] = load_frag(&lA[p][(wm * 32 + 0)  * LDR], LDR, lane);
    afr[1] = load_frag(&lA[p][(wm * 32 + 16) * LDR], LDR, lane);
    bfr[0] = load_frag(&lB[p][(wn * 32 + 0)  * LDR], LDR, lane);
    bfr[1] = load_frag(&lB[p][(wn * 32 + 16) * LDR], LDR, lane);

#pragma unroll
    for (int mt = 0; mt < 2; ++mt)
#pragma unroll
      for (int nt = 0; nt < 2; ++nt)
        acc[mt][nt] = __builtin_amdgcn_wmma_f32_16x16x32_bf16(
            false, afr[mt], false, bfr[nt], (short)0, acc[mt][nt], false, false);

    __syncthreads();
  }

  // ---- epilogue (full tiles -> no bounds checks) ----
  const int rhi = (lane & 16) ? 8 : 0;
  const int cl  = lane & 15;
#pragma unroll
  for (int mt = 0; mt < 2; ++mt)
#pragma unroll
    for (int nt = 0; nt < 2; ++nt)
#pragma unroll
      for (int r = 0; r < 8; ++r) {
        int row = rowbase + wm * 32 + mt * 16 + r + rhi;
        int col = colbase + wn * 32 + nt * 16 + cl;
        float v = acc[mt][nt][r];
        if (bias)    v += bias[col];
        if (do_gelu) v = gelu_exact(v);
        if (resid)   v += resid[(size_t)row * ldc + col];
        if (C32)     C32[(size_t)row * ldc + col] = v;
        if (Cbf)     Cbf[(size_t)row * ldc + col] = f32_to_bf16(v);
        if (CbfT && col >= tc0)
          CbfT[(size_t)(col - tc0) * ldcT + row] = f32_to_bf16(v);
      }
}

// ---------------- fused ReLA attention ----------------
// Per (64-row block, head): O = sum_j relu(causal(q@k^T * DH^-0.5)) @ V
// No S materialization; causal upper triangle never visited (loop bound).
__global__ void __launch_bounds__(GT)
rela_attn_kernel(const unsigned short* __restrict__ qkv,   // [L, 3*INNERD] bf16
                 const unsigned short* __restrict__ vT,    // [INNERD, L]   bf16
                 unsigned short* __restrict__ O) {         // [L, INNERD]   bf16
  constexpr int TS = 64;          // query-rows / key-block / head-dim
  constexpr int SR = TS + 8;      // padded LDS row stride (halves)

  __shared__ __align__(16) unsigned short lQ[TS * SR];
  __shared__ __align__(16) unsigned short lK[TS * SR];
  __shared__ __align__(16) unsigned short lV[TS * SR];
  __shared__ __align__(16) unsigned short lS[TS * SR];

  const int tid  = threadIdx.x;
  const int lane = tid & 31;
  const int wave = tid >> 5;
  const int sm   = wave >> 1;           // 0..3 : 16-row strip
  const int sn0  = (wave & 1) * 2;      // 0|2  : pair of 16-col strips
  const int rowbase = blockIdx.x * TS;
  const int hd      = blockIdx.y;
  const int rhi = (lane & 16) ? 8 : 0;
  const int cl  = lane & 15;

  // ---- stage Q tile once ----
#pragma unroll
  for (int c = 0; c < 2; ++c) {
    int f8 = tid + c * GT;
    int r = f8 >> 3, ko = (f8 & 7) * 8;
    stage16(qkv + hd * DHEAD + (size_t)(rowbase + r) * (3 * INNERD) + ko,
            lQ + r * SR + ko);
  }
  stage_wait_all();
  __syncthreads();

  v16bf qf[2];
#pragma unroll
  for (int ks = 0; ks < 2; ++ks)
    qf[ks] = load_frag(lQ + sm * 16 * SR + ks * 32, SR, lane);

  v8f oacc[2] = {};

  for (int j0 = 0; j0 <= rowbase; j0 += TS) {
    // ---- stage K (rows of k) and V^T tiles ----
#pragma unroll
    for (int c = 0; c < 2; ++c) {
      int f8 = tid + c * GT;
      int r = f8 >> 3, ko = (f8 & 7) * 8;
      stage16(qkv + INNERD + hd * DHEAD + (size_t)(j0 + r) * (3 * INNERD) + ko,
              lK + r * SR + ko);
      stage16(vT + (size_t)(hd * DHEAD + r) * LSEQ + j0 + ko,
              lV + r * SR + ko);
    }
    stage_wait_all();
    __syncthreads();

    // ---- S tile = relu(causal(scale * q @ k^T)) -> lS (bf16) ----
    v8f sacc[2] = {};
#pragma unroll
    for (int t = 0; t < 2; ++t)
#pragma unroll
      for (int ks = 0; ks < 2; ++ks) {
        v16bf kf = load_frag(lK + (sn0 + t) * 16 * SR + ks * 32, SR, lane);
        sacc[t] = __builtin_amdgcn_wmma_f32_16x16x32_bf16(
            false, qf[ks], false, kf, (short)0, sacc[t], false, false);
      }
#pragma unroll
    for (int t = 0; t < 2; ++t)
#pragma unroll
      for (int r = 0; r < 8; ++r) {
        int rs = sm * 16 + r + rhi;
        int cs = (sn0 + t) * 16 + cl;
        float v = sacc[t][r] * 0.125f;                       // DHEAD^-0.5
        v = ((j0 + cs <= rowbase + rs) && v > 0.0f) ? v : 0.0f;
        lS[rs * SR + cs] = f32_to_bf16(v);
      }
    __syncthreads();

    // ---- O += S @ V ----
#pragma unroll
    for (int ks = 0; ks < 2; ++ks) {
      v16bf sf = load_frag(lS + sm * 16 * SR + ks * 32, SR, lane);
#pragma unroll
      for (int t = 0; t < 2; ++t) {
        v16bf vf = load_frag(lV + (sn0 + t) * 16 * SR + ks * 32, SR, lane);
        oacc[t] = __builtin_amdgcn_wmma_f32_16x16x32_bf16(
            false, sf, false, vf, (short)0, oacc[t], false, false);
      }
    }
    __syncthreads();
  }

  // ---- write O (bf16, column block of this head) ----
#pragma unroll
  for (int t = 0; t < 2; ++t)
#pragma unroll
    for (int r = 0; r < 8; ++r) {
      int row = rowbase + sm * 16 + r + rhi;
      int col = hd * DHEAD + (sn0 + t) * 16 + cl;
      O[(size_t)row * INNERD + col] = f32_to_bf16(oacc[t][r]);
    }
}

// ---------------- LayerNorm (row-wise, D=1024) ----------------
__global__ void __launch_bounds__(256)
ln_kernel(const float* __restrict__ in, const float* __restrict__ resid,
          const float* __restrict__ g, const float* __restrict__ b,
          float* __restrict__ out32, unsigned short* __restrict__ outbf) {
  const int row = blockIdx.x;
  const int t = threadIdx.x;
  const float* x = in + (size_t)row * DMODEL;
  __shared__ float red[256];

  float v[4];
  float s = 0.0f;
#pragma unroll
  for (int i = 0; i < 4; ++i) { v[i] = x[t + i * 256]; s += v[i]; }
  red[t] = s; __syncthreads();
  for (int off = 128; off > 0; off >>= 1) {
    if (t < off) red[t] += red[t + off];
    __syncthreads();
  }
  const float mean = red[0] * (1.0f / DMODEL);
  __syncthreads();

  float ss = 0.0f;
#pragma unroll
  for (int i = 0; i < 4; ++i) { float d = v[i] - mean; ss += d * d; }
  red[t] = ss; __syncthreads();
  for (int off = 128; off > 0; off >>= 1) {
    if (t < off) red[t] += red[t + off];
    __syncthreads();
  }
  const float rstd = rsqrtf(red[0] * (1.0f / DMODEL) + 1e-5f);

#pragma unroll
  for (int i = 0; i < 4; ++i) {
    int col = t + i * 256;
    float o = (v[i] - mean) * rstd * g[col] + b[col];
    if (resid) o += resid[(size_t)row * DMODEL + col];
    if (out32) out32[(size_t)row * DMODEL + col] = o;
    if (outbf) outbf[(size_t)row * DMODEL + col] = f32_to_bf16(o);
  }
}

// ---------------- embedding gather ----------------
__global__ void __launch_bounds__(256)
embed_kernel(const int* __restrict__ x, const float* __restrict__ tok,
             const float* __restrict__ pos, float* __restrict__ h) {
  size_t idx = (size_t)blockIdx.x * 256 + threadIdx.x;   // L*D total
  int i = (int)(idx / DMODEL);
  int d = (int)(idx % DMODEL);
  h[idx] = tok[(size_t)x[i] * DMODEL + d] + pos[idx];
}

// ---------------- tiled transpose + f32->bf16: out[c][r] = in[r][c] ----------------
__global__ void __launch_bounds__(256)
transpose_cvt_kernel(const float* __restrict__ in, unsigned short* __restrict__ out,
                     int R, int C) {
  __shared__ float tile[32][33];
  const int c0 = blockIdx.x * 32, r0 = blockIdx.y * 32;
  const int tx = threadIdx.x & 31, ty = threadIdx.x >> 5;   // 32 x 8
#pragma unroll
  for (int i = 0; i < 4; ++i)
    tile[ty + i * 8][tx] = in[(size_t)(r0 + ty + i * 8) * C + c0 + tx];
  __syncthreads();
#pragma unroll
  for (int i = 0; i < 4; ++i)
    out[(size_t)(c0 + ty + i * 8) * R + r0 + tx] = f32_to_bf16(tile[tx][ty + i * 8]);
}

// ---------------- host driver ----------------
extern "C" void kernel_launch(void* const* d_in, const int* in_sizes, int n_in,
                              void* d_out, int out_size, void* d_ws, size_t ws_size,
                              hipStream_t stream) {
  (void)in_sizes; (void)n_in; (void)out_size; (void)ws_size;

  const int*   x       = (const int*)d_in[0];
  const float* tok_emb = (const float*)d_in[1];
  const float* pos_emb = (const float*)d_in[2];
  const float* attn_ng = (const float*)d_in[3];
  const float* attn_nb = (const float*)d_in[4];
  const float* wqkv    = (const float*)d_in[5];
  const float* wo      = (const float*)d_in[6];
  const float* bo      = (const float*)d_in[7];
  const float* out_ng  = (const float*)d_in[8];
  const float* out_nb  = (const float*)d_in[9];
  const float* ff_ng   = (const float*)d_in[10];
  const float* ff_nb   = (const float*)d_in[11];
  const float* w1      = (const float*)d_in[12];
  const float* b1      = (const float*)d_in[13];
  const float* w2      = (const float*)d_in[14];
  const float* b2      = (const float*)d_in[15];
  const float* fin_g   = (const float*)d_in[16];
  const float* fin_b   = (const float*)d_in[17];
  const float* w_logit = (const float*)d_in[18];
  const float* b_logit = (const float*)d_in[19];
  float* logits = (float*)d_out;

  // ---- workspace carve-up ----
  char* ws = (char*)d_ws;
  size_t off = 0;
  auto carve = [&](size_t bytes) -> char* {
    char* p = ws + off;
    off += (bytes + 255) & ~(size_t)255;
    return p;
  };
  float*          h     = (float*)         carve((size_t)LSEQ * DMODEL * 4);      //  8 MB
  unsigned short* ybf   = (unsigned short*)carve((size_t)LSEQ * DMODEL * 2);      //  4 MB
  unsigned short* qkvbf = (unsigned short*)carve((size_t)LSEQ * 3 * INNERD * 2);  // 12 MB
  unsigned short* vT    = (unsigned short*)carve((size_t)INNERD * LSEQ * 2);      //  4 MB
  unsigned short* obf   = (unsigned short*)carve((size_t)LSEQ * INNERD * 2);      //  4 MB
  float*          oproj = (float*)         carve((size_t)LSEQ * DMODEL * 4);      //  8 MB
  unsigned short* ffbf  = (unsigned short*)carve((size_t)LSEQ * FFDIM * 2);       // 16 MB
  unsigned short* wT    = (unsigned short*)carve((size_t)NVOCAB * DMODEL * 2);    // 64 MB

  const dim3 blk(GT);

  auto gemm = [&](const unsigned short* A, int lda, const unsigned short* B, int ldb,
                  float* C32, unsigned short* Cbf, int ldc,
                  unsigned short* CbfT, int ldcT, int tc0,
                  const float* bias, const float* resid,
                  int M, int N, int K, int do_gelu) {
    dim3 grid((unsigned)(N / 128), (unsigned)(M / 64));
    gemm_wmma_kernel<<<grid, blk, 0, stream>>>(
        A, lda, B, ldb, C32, Cbf, ldc, CbfT, ldcT, tc0, bias, resid,
        M, N, K, do_gelu);
  };
  auto tcvt = [&](const float* src, unsigned short* dst, int R, int C) {
    transpose_cvt_kernel<<<dim3(C / 32, R / 32), blk, 0, stream>>>(src, dst, R, C);
  };
  auto ln = [&](const float* in, const float* resid, const float* g, const float* b,
                float* o32, unsigned short* obf16) {
    ln_kernel<<<dim3(LSEQ), blk, 0, stream>>>(in, resid, g, b, o32, obf16);
  };

  // ---- h = tok_emb[x] + pos_emb ----
  embed_kernel<<<dim3((LSEQ * DMODEL) / 256), blk, 0, stream>>>(x, tok_emb, pos_emb, h);

  for (int l = 0; l < NDEPTH; ++l) {
    // === ReLA attention ===
    ln(h, nullptr, attn_ng + l * DMODEL, attn_nb + l * DMODEL, nullptr, ybf);
    tcvt(wqkv + (size_t)l * DMODEL * 3 * INNERD, wT, DMODEL, 3 * INNERD);
    // qkv (peel V^T into vT from the epilogue)
    gemm(ybf, DMODEL, wT, DMODEL, nullptr, qkvbf, 3 * INNERD,
         vT, LSEQ, 2 * INNERD, nullptr, nullptr, LSEQ, 3 * INNERD, DMODEL, 0);

    // fused flash-style ReLA attention: all heads, one launch
    rela_attn_kernel<<<dim3(LSEQ / 64, NHEAD), blk, 0, stream>>>(qkvbf, vT, obf);

    // out-proj + bias, then LN + residual into h
    tcvt(wo + (size_t)l * INNERD * DMODEL, wT, INNERD, DMODEL);
    gemm(obf, INNERD, wT, INNERD, oproj, nullptr, DMODEL, nullptr, 0, 0,
         bo + l * DMODEL, nullptr, LSEQ, DMODEL, INNERD, 0);
    ln(oproj, /*resid=*/h, out_ng + l * DMODEL, out_nb + l * DMODEL, h, nullptr);

    // === FFN ===
    ln(h, nullptr, ff_ng + l * DMODEL, ff_nb + l * DMODEL, nullptr, ybf);
    tcvt(w1 + (size_t)l * DMODEL * FFDIM, wT, DMODEL, FFDIM);
    gemm(ybf, DMODEL, wT, DMODEL, nullptr, ffbf, FFDIM, nullptr, 0, 0,
         b1 + l * FFDIM, nullptr, LSEQ, FFDIM, DMODEL, /*gelu=*/1);
    tcvt(w2 + (size_t)l * FFDIM * DMODEL, wT, FFDIM, DMODEL);
    gemm(ffbf, FFDIM, wT, FFDIM, h, nullptr, DMODEL, nullptr, 0, 0,
         b2 + l * DMODEL, /*resid=*/h, LSEQ, DMODEL, FFDIM, 0);
  }

  // === final LN + logits ===
  ln(h, nullptr, fin_g, fin_b, nullptr, ybf);
  tcvt(w_logit, wT, DMODEL, NVOCAB);
  gemm(ybf, DMODEL, wT, DMODEL, logits, nullptr, NVOCAB, nullptr, 0, 0,
       b_logit, nullptr, LSEQ, NVOCAB, DMODEL, 0);
}
